// CrossAttention_1305670058069
// MI455X (gfx1250) — compile-verified, matrix operand order checked by reading, not run on hardware
//
#include <hip/hip_runtime.h>
#include <hip/hip_bf16.h>

typedef __attribute__((ext_vector_type(16))) _Float16 v16h;
typedef __attribute__((ext_vector_type(8)))  _Float16 v8h;
typedef __attribute__((ext_vector_type(4)))  _Float16 v4h;
typedef __attribute__((ext_vector_type(8)))  float    v8f;

typedef __attribute__((ext_vector_type(4))) unsigned int tdm_v4u;
typedef __attribute__((ext_vector_type(8))) int          tdm_v8i;
typedef __attribute__((ext_vector_type(4))) int          tdm_v4i;

// ---------------------------------------------------------------------------
// Fragment load for v_wmma_f32_16x16x32_f16 A/B operands.
// lane L (l = L&15, hi = L>>4) holds row/col l:
//   halves 0..7  -> K = hi*8 + 0..7        (16-byte contiguous run)
//   halves 8..15 -> K = hi*8 + 16 .. +23   (second run)
// -> two ds_load_b128 per fragment when row stride is a multiple of 8 halves.
// ---------------------------------------------------------------------------
__device__ __forceinline__ v16h load_frag(const _Float16* row, int hi) {
  union { v16h v; v8h h[2]; } u;
  u.h[0] = *(const v8h*)(row + hi * 8);
  u.h[1] = *(const v8h*)(row + hi * 8 + 16);
  return u.v;
}

__device__ __forceinline__ v8f wmma_f16(v16h a, v16h b, v8f c) {
  return __builtin_amdgcn_wmma_f32_16x16x32_f16(false, a, false, b, (short)0, c,
                                                false, false);
}

// ---------------------------------------------------------------------------
// Tensor Data Mover: 1-D contiguous copy of nelem f16 elements into LDS.
// D# per CDNA5 ISA 8.3/8.4: group0 = {count=1 | lds_addr | global_addr | type=2},
// group1 = {data_size=2B, tensor_dim0 = tile_dim0 = nelem, dims1/2 = 1}.
// This toolchain exposes the 6-arg builtin:
//   (uint32x4 g0, int32x8 g1, int32x4 g2, int32x4 g3, int32x8 extra, i32 cpol)
// ---------------------------------------------------------------------------
#if __has_builtin(__builtin_amdgcn_tensor_load_to_lds)
#define HAVE_TDM 1
__device__ __forceinline__ void tdm_load_lds_1d(_Float16* lds, const _Float16* g,
                                                unsigned nelem) {
  unsigned lds_off = (unsigned)(unsigned long long)lds;  // LDS addr = generic[31:0]
  unsigned long long ga = (unsigned long long)g;
  tdm_v4u g0;
  g0[0] = 1u;                                              // count=1 (valid), user mode
  g0[1] = lds_off;                                         // lds_addr
  g0[2] = (unsigned)ga;                                    // global_addr[31:0]
  g0[3] = (unsigned)((ga >> 32) & 0x01ffffffu) | 0x80000000u;  // addr[56:32] | type=2
  tdm_v8i g1;
  g1[0] = (int)0x00010000u;                                // data_size = 1 -> 2 bytes
  g1[1] = (int)((nelem & 0xffffu) << 16);                  // tensor_dim0[15:0]
  g1[2] = (int)(((nelem >> 16) & 0xffffu) | (1u << 16));   // dim0 hi | tensor_dim1 = 1
  g1[3] = (int)((nelem & 0xffffu) << 16);                  // dim1 hi(0) | tile_dim0
  g1[4] = (int)((1u) | (1u << 16));                        // tile_dim1 = 1, tile_dim2 = 1
  g1[5] = (int)nelem;                                      // tensor_dim0_stride lo
  g1[6] = 0;
  g1[7] = 0;
  tdm_v4i z4 = {0, 0, 0, 0};
  tdm_v8i z8 = {0, 0, 0, 0, 0, 0, 0, 0};
  __builtin_amdgcn_tensor_load_to_lds(g0, g1, z4, z4, z8, 0);
}
#else
#define HAVE_TDM 0
#endif

// ---------------------------------------------------------------------------
// Small helper kernels
// ---------------------------------------------------------------------------
__global__ void f32_to_f16_v4(const float* __restrict__ in, _Float16* __restrict__ out,
                              int n4) {
  int i = blockIdx.x * 256 + threadIdx.x;
  if (i < n4) {
    float4 v = ((const float4*)in)[i];
    v4h o = {(_Float16)v.x, (_Float16)v.y, (_Float16)v.z, (_Float16)v.w};
    ((v4h*)out)[i] = o;
  }
}

__global__ void zero_f16(_Float16* __restrict__ p, int n) {
  int i = blockIdx.x * 256 + threadIdx.x;
  if (i < n) p[i] = (_Float16)0.0f;
}

// ---------------------------------------------------------------------------
// Double-buffered WMMA GEMM: C[M,N] = A[M,K](f16) * B[K,N](f16) (+bias)
// OUT_F32 == 0: f16 output, no bias.   OUT_F32 == 1: f32 output + bias.
// Block tile 128x128, K-step 32, 8 waves; wave w owns rows [w*16, w*16+16),
// 8 WMMAs per wave per K-step. Global loads for step t+1 are issued before
// the WMMAs of step t; one barrier per step.
// ---------------------------------------------------------------------------
#define BM 128
#define BN 128
#define BK 32
#define LDA 40  // LDS row stride in halves (fragment runs stay 16B aligned)

template <int OUT_F32>
__global__ __launch_bounds__(256) void gemm_wmma(const _Float16* __restrict__ A,
                                                 const _Float16* __restrict__ Bw,
                                                 const float* __restrict__ bias,
                                                 void* __restrict__ Outp,
                                                 int Kd, int Nc) {
  __shared__ __align__(16) _Float16 As[2][BM * LDA];
  __shared__ __align__(16) _Float16 Bs[2][BN * LDA];
  const int tid  = threadIdx.x;
  const int w    = tid >> 5;
  const int lane = tid & 31;
  const int l    = lane & 15;
  const int hi   = lane >> 4;
  const int m0   = blockIdx.x * BM;
  const int n0   = blockIdx.y * BN;

  // staging assignments: A row ra, 16 halves at c0; B row kb, 16 halves at nb0
  const int ra  = tid >> 1;
  const int c0  = (tid & 1) * 16;
  const int kb  = tid >> 3;
  const int nb0 = (tid & 7) * 16;

  v8h a0, a1, b0, b1;
  auto loadG = [&](int k0) {
    const v8h* pa = (const v8h*)&A[(size_t)(m0 + ra) * Kd + k0 + c0];
    a0 = pa[0];
    a1 = pa[1];
    const v8h* pb = (const v8h*)&Bw[(size_t)(k0 + kb) * Nc + n0 + nb0];
    b0 = pb[0];
    b1 = pb[1];
  };
  auto storeL = [&](int buf) {
    v8h* qa = (v8h*)&As[buf][ra * LDA + c0];
    qa[0] = a0;
    qa[1] = a1;
    union { v8h v; _Float16 s[8]; } u0, u1;
    u0.v = b0;
    u1.v = b1;
    for (int j = 0; j < 8; ++j) {
      Bs[buf][(nb0 + j) * LDA + kb]     = u0.s[j];
      Bs[buf][(nb0 + 8 + j) * LDA + kb] = u1.s[j];
    }
  };

  v8f acc[8];
  for (int s = 0; s < 8; ++s)
    for (int r = 0; r < 8; ++r) acc[s][r] = 0.0f;

  loadG(0);
  storeL(0);
  __syncthreads();

  const int nT = Kd / BK;
  for (int t = 0; t < nT; ++t) {
    const int cur = t & 1;
    if (t + 1 < nT) loadG((t + 1) * BK);          // prefetch next tile into regs
    v16h af = load_frag(&As[cur][(w * 16 + l) * LDA], hi);
    for (int s = 0; s < 8; ++s) {
      v16h bf = load_frag(&Bs[cur][(s * 16 + l) * LDA], hi);
      acc[s] = wmma_f16(af, bf, acc[s]);
    }
    if (t + 1 < nT) storeL(cur ^ 1);              // fill the other buffer
    __syncthreads();
  }

  if (OUT_F32 == 0) {
    _Float16* Out = (_Float16*)Outp;
    for (int s = 0; s < 8; ++s)
      for (int r = 0; r < 8; ++r) {
        int gmr = m0 + w * 16 + r + hi * 8;
        int gnc = n0 + s * 16 + l;
        Out[(size_t)gmr * Nc + gnc] = (_Float16)acc[s][r];
      }
  } else {
    float* Out = (float*)Outp;
    for (int s = 0; s < 8; ++s)
      for (int r = 0; r < 8; ++r) {
        int gmr = m0 + w * 16 + r + hi * 8;
        int gnc = n0 + s * 16 + l;
        Out[(size_t)gmr * Nc + gnc] = acc[s][r] + bias[gnc];
      }
  }
}

// ---------------------------------------------------------------------------
// K/V projection: context[b,77,768] @ Wk/Wv[768,640] -> padded f16 buffers.
//   Kbuf[b][h][m(96)][d(96)]   (rows 77..95 / cols 80..95 zero)
//   Vbuf[b][h][d(96)][m(96)]   (transposed for contiguous B-fragments)
// grid = (ntile=10, rowtile=5, b=16); waves 0-3 -> K, waves 4-7 -> V.
// ---------------------------------------------------------------------------
__global__ __launch_bounds__(256) void kv_proj(const float* __restrict__ ctx,
                                               const _Float16* __restrict__ Wk16,
                                               const _Float16* __restrict__ Wv16,
                                               _Float16* __restrict__ Kbuf,
                                               _Float16* __restrict__ Vbuf) {
  __shared__ __align__(16) _Float16 As[16 * LDA];
  __shared__ __align__(16) _Float16 Bks[64 * LDA];
  __shared__ __align__(16) _Float16 Bvs[64 * LDA];
  const int b  = blockIdx.z;
  const int rt = blockIdx.y;
  const int n0 = blockIdx.x * 64;
  const int tid = threadIdx.x, w = tid >> 5, lane = tid & 31;
  const int l = lane & 15, hi = lane >> 4;
  const int sub = w & 3;

  v8f acc;
  for (int r = 0; r < 8; ++r) acc[r] = 0.0f;

  for (int k0 = 0; k0 < 768; k0 += 32) {
    for (int e = tid; e < 16 * 32; e += 256) {
      int r = e >> 5, c = e & 31;
      int m = rt * 16 + r;
      As[r * LDA + c] =
          (m < 77) ? (_Float16)ctx[((size_t)b * 77 + m) * 768 + k0 + c] : (_Float16)0.0f;
    }
    for (int e = tid; e < 64 * 32; e += 256) {
      int k = e >> 6, n = e & 63;
      Bks[n * LDA + k] = Wk16[(size_t)(k0 + k) * 640 + n0 + n];
      Bvs[n * LDA + k] = Wv16[(size_t)(k0 + k) * 640 + n0 + n];
    }
    __syncthreads();
    const _Float16* Bsrc = (w < 4) ? Bks : Bvs;
    v16h af = load_frag(&As[l * LDA], hi);
    v16h bf = load_frag(&Bsrc[(sub * 16 + l) * LDA], hi);
    acc = wmma_f16(af, bf, acc);
    __syncthreads();
  }

  int c  = n0 + sub * 16 + l;  // 16-col subtile never crosses a head (80 % 16 == 0)
  int hh = c / 80, dh = c % 80;
  for (int r = 0; r < 8; ++r) {
    int m = rt * 16 + r + hi * 8;  // 0..79 (77..79 compute exact zeros)
    _Float16 v = (_Float16)acc[r];
    if (w < 4) Kbuf[(((size_t)b * 8 + hh) * 96 + m) * 96 + dh] = v;
    else       Vbuf[(((size_t)b * 8 + hh) * 96 + dh) * 96 + m] = v;
  }
}

// ---------------------------------------------------------------------------
// Fused attention: block = (128 queries, head, batch); 8 waves x 16 queries.
// K/V staged via the Tensor Data Mover (s_wait_tensorcnt), scores & PV via
// v_wmma_f32_16x16x32_f16, softmax via shuffle reductions over the C-fragment
// lane groups. Layout guidance: key col 0 exempt, cols >= 77 masked.
// ---------------------------------------------------------------------------
__global__ __launch_bounds__(256) void attn_kernel(const _Float16* __restrict__ Q16,
                                                   const _Float16* __restrict__ Kbuf,
                                                   const _Float16* __restrict__ Vbuf,
                                                   const float* __restrict__ gmask,
                                                   _Float16* __restrict__ O16) {
  __shared__ __align__(16) _Float16 Klds[96 * 96];      // [m][d]
  __shared__ __align__(16) _Float16 Vlds[96 * 96];      // [d][m]
  __shared__ __align__(16) _Float16 Plds[8][16 * 96];   // per-wave Q staging, then P
  const int b  = blockIdx.z;
  const int h  = blockIdx.y;
  const int q0 = blockIdx.x * 128;
  const int tid = threadIdx.x, w = tid >> 5, lane = tid & 31;
  const int l = lane & 15, hi = lane >> 4;

  const _Float16* Kg = Kbuf + ((size_t)(b * 8 + h)) * 96 * 96;
  const _Float16* Vg = Vbuf + ((size_t)(b * 8 + h)) * 96 * 96;
#if HAVE_TDM
  if (w == 0) {  // one wave drives the DMA; TENSORcnt is per-wave
    tdm_load_lds_1d(Klds, Kg, 96 * 96);
    tdm_load_lds_1d(Vlds, Vg, 96 * 96);
    __builtin_amdgcn_s_wait_tensorcnt(0);
  }
#else
  {
    const uint4* Kg4 = (const uint4*)Kg;
    const uint4* Vg4 = (const uint4*)Vg;
    uint4* Kl = (uint4*)Klds;
    uint4* Vl = (uint4*)Vlds;
    for (int e = tid; e < 96 * 96 / 8; e += 256) { Kl[e] = Kg4[e]; Vl[e] = Vg4[e]; }
  }
#endif
  // stage this wave's 16 Q rows, zero-padded d=80..95 (padding later reused by P)
  _Float16* Pw = Plds[w];
  {
    const _Float16* Qg = Q16 + ((size_t)b * 4096 + q0 + w * 16) * 640 + (size_t)h * 80;
    for (int e = lane; e < 16 * 96; e += 32) {
      int i = e / 96, c = e - i * 96;
      Pw[i * 96 + c] = (c < 80) ? Qg[(size_t)i * 640 + c] : (_Float16)0.0f;
    }
  }
  __syncthreads();

  // ---- scores: sim[16][80] over d = 0..95 (zero padded) ----
  v8f accS[5];
  for (int mt = 0; mt < 5; ++mt)
    for (int r = 0; r < 8; ++r) accS[mt][r] = 0.0f;
  for (int ks = 0; ks < 3; ++ks) {
    int d0 = ks * 32;
    v16h af = load_frag(&Pw[l * 96 + d0], hi);
    for (int mt = 0; mt < 5; ++mt) {
      v16h bf = load_frag(&Klds[(mt * 16 + l) * 96 + d0], hi);
      accS[mt] = wmma_f16(af, bf, accS[mt]);
    }
  }

  // ---- scale + layout guidance + key mask ----
  float gmv[8];
  for (int r = 0; r < 8; ++r) {
    int n = q0 + w * 16 + r + hi * 8;
    float g = gmask[(size_t)b * 4096 + n];
    gmv[r] = (g == 1.0f) ? 5.0f : ((g == 0.0f) ? 0.1f : g);
  }
  const float SCALE = 0.11180339887498948f;  // 80^-0.5
  float sv[5][8];
  for (int mt = 0; mt < 5; ++mt) {
    int m = mt * 16 + l;
    for (int r = 0; r < 8; ++r) {
      float vq = accS[mt][r] * SCALE;
      if (m >= 1) vq *= gmv[r];
      if (m >= 77) vq = -3.0e38f;  // pad keys -> exp() == 0
      sv[mt][r] = vq;
    }
  }

  // ---- softmax per row (row spans the 16-lane group of the C fragment) ----
  for (int r = 0; r < 8; ++r) {
    float mx = sv[0][r];
    for (int mt = 1; mt < 5; ++mt) mx = fmaxf(mx, sv[mt][r]);
    for (int off = 8; off >= 1; off >>= 1) mx = fmaxf(mx, __shfl_xor(mx, off, 16));
    float s = 0.0f;
    for (int mt = 0; mt < 5; ++mt) {
      float e = __expf(sv[mt][r] - mx);
      sv[mt][r] = e;
      s += e;
    }
    for (int off = 8; off >= 1; off >>= 1) s += __shfl_xor(s, off, 16);
    float inv = 1.0f / s;
    for (int mt = 0; mt < 5; ++mt) sv[mt][r] *= inv;
  }

  // ---- write P into A-fragment layout (cols 80..95 stay zero from Q staging) ----
  for (int mt = 0; mt < 5; ++mt)
    for (int r = 0; r < 8; ++r)
      Pw[(r + hi * 8) * 96 + mt * 16 + l] = (_Float16)sv[mt][r];
  // same-wave LDS ops are in order; compiler inserts the dscnt waits

  // ---- O = P V over m = 0..95 (P and V zero for m >= 77) ----
  v8f accO[5];
  for (int dt = 0; dt < 5; ++dt)
    for (int r = 0; r < 8; ++r) accO[dt][r] = 0.0f;
  for (int ks = 0; ks < 3; ++ks) {
    int m0 = ks * 32;
    v16h af = load_frag(&Pw[l * 96 + m0], hi);
    for (int dt = 0; dt < 5; ++dt) {
      v16h bf = load_frag(&Vlds[(dt * 16 + l) * 96 + m0], hi);
      accO[dt] = wmma_f16(af, bf, accO[dt]);
    }
  }

  _Float16* Og = O16 + ((size_t)b * 4096 + q0 + w * 16) * 640 + (size_t)h * 80;
  for (int dt = 0; dt < 5; ++dt)
    for (int r = 0; r < 8; ++r)
      Og[(size_t)(r + hi * 8) * 640 + dt * 16 + l] = (_Float16)accO[dt][r];
}

// ---------------------------------------------------------------------------
// Host launch
// ---------------------------------------------------------------------------
extern "C" void kernel_launch(void* const* d_in, const int* in_sizes, int n_in,
                              void* d_out, int out_size, void* d_ws, size_t ws_size,
                              hipStream_t stream) {
  (void)in_sizes; (void)n_in; (void)out_size; (void)ws_size;
  const float* x    = (const float*)d_in[0];
  const float* ctx  = (const float*)d_in[1];
  const float* gm   = (const float*)d_in[2];
  const float* Wq   = (const float*)d_in[3];
  const float* Wk   = (const float*)d_in[4];
  const float* Wv   = (const float*)d_in[5];
  const float* Wout = (const float*)d_in[6];
  const float* bout = (const float*)d_in[7];

  char* ws = (char*)d_ws;
  size_t off = 0;
  auto take = [&](size_t bytes) -> void* {
    void* p = ws + off;
    off += (bytes + 255) & ~(size_t)255;
    return p;
  };
  const size_t NQ = (size_t)65536 * 640;  // B*N rows x INNER/QD cols
  _Float16* X16  = (_Float16*)take(NQ * 2);
  _Float16* Q16  = (_Float16*)take(NQ * 2);
  _Float16* O16  = (_Float16*)take(NQ * 2);
  _Float16* Wq16 = (_Float16*)take((size_t)640 * 640 * 2);
  _Float16* Wk16 = (_Float16*)take((size_t)768 * 640 * 2);
  _Float16* Wv16 = (_Float16*)take((size_t)768 * 640 * 2);
  _Float16* Wo16 = (_Float16*)take((size_t)640 * 640 * 2);
  _Float16* Kbuf = (_Float16*)take((size_t)16 * 8 * 96 * 96 * 2);
  _Float16* Vbuf = (_Float16*)take((size_t)16 * 8 * 96 * 96 * 2);

  f32_to_f16_v4<<<(int)((NQ / 4 + 255) / 256), 256, 0, stream>>>(x, X16, (int)(NQ / 4));
  f32_to_f16_v4<<<(640 * 640 / 4 + 255) / 256, 256, 0, stream>>>(Wq, Wq16, 640 * 640 / 4);
  f32_to_f16_v4<<<(768 * 640 / 4 + 255) / 256, 256, 0, stream>>>(Wk, Wk16, 768 * 640 / 4);
  f32_to_f16_v4<<<(768 * 640 / 4 + 255) / 256, 256, 0, stream>>>(Wv, Wv16, 768 * 640 / 4);
  f32_to_f16_v4<<<(640 * 640 / 4 + 255) / 256, 256, 0, stream>>>(Wout, Wo16, 640 * 640 / 4);

  const int nkv = 16 * 8 * 96 * 96;
  zero_f16<<<(nkv + 255) / 256, 256, 0, stream>>>(Kbuf, nkv);
  zero_f16<<<(nkv + 255) / 256, 256, 0, stream>>>(Vbuf, nkv);

  // Q = x @ Wq  (f16 out)
  gemm_wmma<0><<<dim3(65536 / BM, 640 / BN), 256, 0, stream>>>(
      X16, Wq16, nullptr, (void*)Q16, 640, 640);

  // K/V projections into padded buffers
  kv_proj<<<dim3(10, 5, 16), 256, 0, stream>>>(ctx, Wk16, Wv16, Kbuf, Vbuf);

  // fused attention
  attn_kernel<<<dim3(4096 / 128, 8, 16), 256, 0, stream>>>(Q16, Kbuf, Vbuf, gm, O16);

  // out = O @ Wout + bout  (f32 out)
  gemm_wmma<1><<<dim3(65536 / BM, 640 / BN), 256, 0, stream>>>(
      O16, Wo16, bout, d_out, 640, 640);
}
